// IntentionBlock_2774548873500
// MI455X (gfx1250) — matrix operand order repaired and checked by reading, not compiled, not run on hardware
//
#include <hip/hip_runtime.h>
#include <hip/hip_bf16.h>

#define B_N  4096
#define D_N  2048
#define K_N  8
#define H2_N 64

typedef __attribute__((ext_vector_type(16))) __bf16 v16bf;
typedef __attribute__((ext_vector_type(2)))  __bf16 v2bf;
typedef __attribute__((ext_vector_type(8)))  float  v8f;

// pack two floats into one dword of two bf16 (RNE). Written as a clean
// build_vector of fptruncs so the backend matches v_cvt_pk_bf16_f32.
__device__ __forceinline__ unsigned pkbf(float a, float b) {
  v2bf r;
  r.x = (__bf16)a;
  r.y = (__bf16)b;
  return __builtin_bit_cast(unsigned, r);
}

// ---------------------------------------------------------------------------
// Kernel 1: bucket samples by one-hot intention; init out[b] with the fused
// fc2 + relu + W3-tail term:  out[b] = b3 + sum_h relu(W2[h,k]+b2[h])*W3[D+h]
// ---------------------------------------------------------------------------
__global__ void prep_kernel(const float* __restrict__ intention,
                            const float* __restrict__ W2,
                            const float* __restrict__ b2,
                            const float* __restrict__ W3,
                            const float* __restrict__ b3,
                            int* __restrict__ count,
                            int* __restrict__ perm,
                            float* __restrict__ out) {
  int b = blockIdx.x * blockDim.x + threadIdx.x;
  if (b >= B_N) return;
  int k = 0;
#pragma unroll
  for (int j = 1; j < K_N; ++j)
    if (intention[b * K_N + j] > 0.5f) k = j;
  int pos = atomicAdd(&count[k], 1);
  perm[k * B_N + pos] = b;

  float acc = b3[0];
#pragma unroll 8
  for (int h = 0; h < H2_N; ++h) {
    float v = W2[h * K_N + k] + b2[h];
    if (v > 0.0f) acc += v * W3[D_N + h];
  }
  out[b] = acc;
}

// ---------------------------------------------------------------------------
// Kernel 2: per-intention-group GEMM with bf16 WMMA, fused bias+relu+W3 dot.
// Block tile: 128 gathered rows x 128 d-columns, depth chunks of 32, double-
// buffered LDS (one barrier per step, global loads overlapped with WMMA).
// 8 waves; wave w owns rows [16w,16w+16) and loops 8 n-subtiles (acc[8]).
// Emits one partial per (sample, 128-wide d block) -> deterministic.
// ---------------------------------------------------------------------------
__global__ __launch_bounds__(256)
void gemm_kernel(const float* __restrict__ x,
                 const float* __restrict__ W1,
                 const float* __restrict__ b1,
                 const float* __restrict__ W3,
                 const int* __restrict__ count,
                 const int* __restrict__ perm,
                 float* __restrict__ partial) {
  const int kk   = blockIdx.z;
  const int cnt  = count[kk];
  const int row0 = blockIdx.x * 128;
  if (row0 >= cnt) return;                 // early-exit empty row tiles
  const int d0 = blockIdx.y * 128;

  __shared__ unsigned short sA[2][128][40];  // [buf][row][depth] bf16
  __shared__ unsigned short sB[2][128][40];  // [buf][d-col][depth] bf16

  const int tid  = threadIdx.x;
  const int lane = tid & 31;
  const int wave = tid >> 5;

  // staging assignment: thread -> (row 0..31 (+32p), 4-float column group)
  const int rloc = tid >> 3;               // 0..31
  const int cg   = (tid & 7) * 4;          // 0,4,...,28

  int samp[4];
#pragma unroll
  for (int p = 0; p < 4; ++p) {
    int gr = row0 + rloc + 32 * p;
    samp[p] = (gr < cnt) ? perm[kk * B_N + gr] : -1;
  }

  v8f acc[8] = {};

  // fragment lane mapping (CDNA5 wave32 WMMA layouts)
  const int mA  = lane & 15;               // A: M = lane%16
  const int kbA = ((lane >> 4) & 1) * 8;   // A: lanes 16-31 hold K+8 groups
  const int kbB = ((lane >> 4) & 1) * 8;   // B: lanes 16-31 hold K+8 groups (same as A)
  const int nl  = lane & 15;               // B/C/D: N = lane%16

  float4 fA[4], fB[4];

  auto ld_tile = [&](int c0) {
#pragma unroll
    for (int p = 0; p < 4; ++p) {
      fA[p] = make_float4(0.f, 0.f, 0.f, 0.f);
      if (samp[p] >= 0)
        fA[p] = *(const float4*)(x + (size_t)samp[p] * D_N + c0 + cg);
    }
#pragma unroll
    for (int p = 0; p < 4; ++p) {
      int dd = rloc + 32 * p;
      fB[p] = *(const float4*)(W1 + ((size_t)(d0 + dd) * K_N + kk) * D_N + c0 + cg);
    }
  };
  auto st_tile = [&](int buf) {
#pragma unroll
    for (int p = 0; p < 4; ++p) {
      int r = rloc + 32 * p;
      *(unsigned*)&sA[buf][r][cg]     = pkbf(fA[p].x, fA[p].y);
      *(unsigned*)&sA[buf][r][cg + 2] = pkbf(fA[p].z, fA[p].w);
      *(unsigned*)&sB[buf][r][cg]     = pkbf(fB[p].x, fB[p].y);
      *(unsigned*)&sB[buf][r][cg + 2] = pkbf(fB[p].z, fB[p].w);
    }
  };

  ld_tile(0);
  st_tile(0);
  int buf = 0;

  for (int c0 = 0; c0 < D_N; c0 += 32) {
    __syncthreads();                       // staged buffer `buf` is ready
    const bool more = (c0 + 32) < D_N;
    if (more) ld_tile(c0 + 32);            // overlap global loads with WMMA

    // A fragment: 16x32 bf16; lanes0-15: K {0-7,16-23}, lanes16-31: +8
    union { v16bf v; unsigned u[8]; } a;
#pragma unroll
    for (int j = 0; j < 8; ++j) {
      int k0 = ((j & 4) << 2) + ((j & 3) << 1) + kbA;
      a.u[j] = *(const unsigned*)&sA[buf][wave * 16 + mA][k0];
    }
#pragma unroll
    for (int ns = 0; ns < 8; ++ns) {
      union { v16bf v; unsigned u[8]; } bb;
#pragma unroll
      for (int j = 0; j < 8; ++j)
        bb.u[j] = *(const unsigned*)&sB[buf][ns * 16 + nl][((j & 4) << 2) + ((j & 3) << 1) + kbB];
      acc[ns] = __builtin_amdgcn_wmma_f32_16x16x32_bf16(
          /*neg_a=*/false, a.v, /*neg_b=*/false, bb.v,
          /*c_mod=*/(short)0, acc[ns], /*reuse_a=*/false, /*reuse_b=*/false);
    }

    if (more) st_tile(buf ^ 1);            // fill the other buffer
    buf ^= 1;
  }

  // ---- epilogue: + b1[d*K+k], relu, * W3[d]; accumulate per-lane across the
  // 8 n-subtiles first, then ONE 16-lane xor-tree reduction per row ----
  float rs[8];
#pragma unroll
  for (int v = 0; v < 8; ++v) rs[v] = 0.f;

#pragma unroll
  for (int ns = 0; ns < 8; ++ns) {
    int d = d0 + ns * 16 + nl;
    float w3d  = W3[d];
    float bias = b1[d * K_N + kk];
#pragma unroll
    for (int v = 0; v < 8; ++v) {
      float val = acc[ns][v] + bias;
      val = val > 0.f ? val : 0.f;
      rs[v] += val * w3d;                  // per-lane partial over this lane's d's
    }
  }
#pragma unroll
  for (int v = 0; v < 8; ++v) {            // xor masks <16 stay in each half
    rs[v] += __shfl_xor(rs[v], 1);
    rs[v] += __shfl_xor(rs[v], 2);
    rs[v] += __shfl_xor(rs[v], 4);
    rs[v] += __shfl_xor(rs[v], 8);
  }

  // lanes 0 and 16 hold reduced sums for rows v and v+8 respectively
  if ((lane & 15) == 0) {
    int half = (lane >> 4) * 8;
#pragma unroll
    for (int v = 0; v < 8; ++v) {
      int gr = row0 + wave * 16 + v + half;
      if (gr < cnt) {
        int b = perm[kk * B_N + gr];
        partial[(size_t)b * 16 + blockIdx.y] = rs[v];
      }
    }
  }
}

// ---------------------------------------------------------------------------
// Kernel 3: fixed-order reduction of the 16 d-block partials per sample.
// ---------------------------------------------------------------------------
__global__ void finalize_kernel(const float* __restrict__ partial,
                                float* __restrict__ out) {
  int b = blockIdx.x * blockDim.x + threadIdx.x;
  if (b >= B_N) return;
  float s = out[b];
  const float* p = partial + (size_t)b * 16;
#pragma unroll
  for (int i = 0; i < 16; ++i) s += p[i];
  out[b] = s;
}

extern "C" void kernel_launch(void* const* d_in, const int* in_sizes, int n_in,
                              void* d_out, int out_size, void* d_ws, size_t ws_size,
                              hipStream_t stream) {
  const float* x         = (const float*)d_in[0];
  const float* intention = (const float*)d_in[1];
  const float* W1        = (const float*)d_in[2];
  const float* b1        = (const float*)d_in[3];
  const float* W2        = (const float*)d_in[4];
  const float* b2        = (const float*)d_in[5];
  const float* W3        = (const float*)d_in[6];
  const float* b3        = (const float*)d_in[7];
  float* out = (float*)d_out;

  char* ws = (char*)d_ws;
  int*   count   = (int*)ws;                                             // 8 ints
  int*   perm    = (int*)(ws + 256);                                     // K*B ints
  float* partial = (float*)(ws + 256 + (size_t)K_N * B_N * sizeof(int)); // B*16 f32

  (void)hipMemsetAsync(count, 0, K_N * sizeof(int), stream);

  prep_kernel<<<B_N / 256, 256, 0, stream>>>(intention, W2, b2, W3, b3,
                                             count, perm, out);

  dim3 g2(B_N / 128, D_N / 128, K_N);
  gemm_kernel<<<g2, 256, 0, stream>>>(x, W1, b1, W3, count, perm, partial);

  finalize_kernel<<<B_N / 256, 256, 0, stream>>>(partial, out);
}
